// Model_1906965479431
// MI455X (gfx1250) — matrix operand order repaired and checked by reading, not compiled, not run on hardware
//
#include <hip/hip_runtime.h>
#include <hip/hip_bf16.h>
#include <math.h>

typedef __bf16 v16bf __attribute__((ext_vector_type(16)));
typedef __bf16 bf16x8 __attribute__((ext_vector_type(8)));
typedef __bf16 bf16x4 __attribute__((ext_vector_type(4)));
typedef float  v8f   __attribute__((ext_vector_type(8)));

union V16 { v16bf v; bf16x8 h[2]; };

#define N_RNA 100000
#define N_DIS 10000
#define FDIM  128
#define HDIM  128
#define NEDGE 1600000
#define NLAB  200000

// ---------------- workspace layout (byte offsets, all 16B aligned) ----------------
constexpr size_t B_AGG_DIS  = 0;                                       // f32 [N_DIS,128]
constexpr size_t B_CNT_DIS  = B_AGG_DIS  + (size_t)N_DIS * FDIM * 4;   // f32 [N_DIS]
constexpr size_t B_AGG_RNA  = B_CNT_DIS  + (size_t)N_DIS * 4;          // f32 [N_RNA,128]
constexpr size_t B_CNT_RNA  = B_AGG_RNA  + (size_t)N_RNA * FDIM * 4;   // f32 [N_RNA]
constexpr size_t B_ZERO_END = B_CNT_RNA  + (size_t)N_RNA * 4;          // zeroed per layer
constexpr size_t B_XB_RNA   = B_ZERO_END;                              // bf16 [N_RNA,128]
constexpr size_t B_XB_DIS   = B_XB_RNA   + (size_t)N_RNA * FDIM * 2;
constexpr size_t B_AGGB_DIS = B_XB_DIS   + (size_t)N_DIS * FDIM * 2;   // bf16 mean(agg)
constexpr size_t B_AGGB_RNA = B_AGGB_DIS + (size_t)N_DIS * FDIM * 2;
constexpr size_t B_H_DIS    = B_AGGB_RNA + (size_t)N_RNA * FDIM * 2;   // bf16 hidden
constexpr size_t B_H_RNA    = B_H_DIS    + (size_t)N_DIS * HDIM * 2;
constexpr size_t B_Z_DIS    = B_H_RNA    + (size_t)N_RNA * HDIM * 2;   // bf16 layer-2 out
constexpr size_t B_Z_RNA    = B_Z_DIS    + (size_t)N_DIS * HDIM * 2;
constexpr size_t B_WT       = B_Z_RNA    + (size_t)N_RNA * HDIM * 2;   // 8x bf16 [128,128] transposed
constexpr size_t B_DW1T     = B_WT       + (size_t)8 * 128 * 128 * 2;  // bf16 [128,256] transposed

// ---------------- zero fill ----------------
__global__ void zero_kernel(float* __restrict__ p, int n) {
    int i = blockIdx.x * blockDim.x + threadIdx.x;
    int stride = gridDim.x * blockDim.x;
    for (; i < n; i += stride) p[i] = 0.0f;
}

// ---------------- f32 -> bf16 copy ----------------
__global__ void cvt_bf16_kernel(const float* __restrict__ in, __bf16* __restrict__ out, int n) {
    int i = blockIdx.x * blockDim.x + threadIdx.x;
    int stride = gridDim.x * blockDim.x;
    for (; i < n; i += stride) out[i] = (__bf16)in[i];
}

// ---------------- f32 [K,128] row-major -> bf16 [128,K] (transposed) ----------------
__global__ void transpose_cvt_kernel(const float* __restrict__ in, __bf16* __restrict__ out, int K) {
    int i = blockIdx.x * blockDim.x + threadIdx.x;
    int stride = gridDim.x * blockDim.x;
    int n = K * 128;
    for (; i < n; i += stride) {
        int k = i >> 7;
        int c = i & 127;
        out[c * K + k] = (__bf16)in[i];
    }
}

// ---------------- scatter-add (one wave32 per edge; bf16 reads, f32 atomic adds) ----------
__global__ __launch_bounds__(256)
void scatter_kernel(const __bf16* __restrict__ xsrc,
                    const int* __restrict__ src,
                    const int* __restrict__ dst,
                    float* __restrict__ agg,
                    float* __restrict__ cnt,
                    int nedges) {
    int wave = (int)((blockIdx.x * blockDim.x + threadIdx.x) >> 5);
    int lane = threadIdx.x & 31;
    if (wave >= nedges) return;
    int s = src[wave];
    int d = dst[wave];
    const bf16x4 v = *(const bf16x4*)(xsrc + (size_t)s * FDIM + lane * 4);
    float* a = agg + (size_t)d * FDIM + lane * 4;
    unsafeAtomicAdd(a + 0, (float)v[0]);
    unsafeAtomicAdd(a + 1, (float)v[1]);
    unsafeAtomicAdd(a + 2, (float)v[2]);
    unsafeAtomicAdd(a + 3, (float)v[3]);
    if (lane == 0) unsafeAtomicAdd(cnt + d, 1.0f);
}

// ---------------- mean finalize: aggb = bf16(agg / max(cnt,1)) ----------------
__global__ void finalize_mean_kernel(const float* __restrict__ agg,
                                     const float* __restrict__ cnt,
                                     __bf16* __restrict__ aggb, int n) {
    int i = blockIdx.x * blockDim.x + threadIdx.x;
    int stride = gridDim.x * blockDim.x;
    for (; i < n; i += stride) {
        int row = i >> 7;
        float inv = 1.0f / fmaxf(cnt[row], 1.0f);
        aggb[i] = (__bf16)(agg[i] * inv);
    }
}

// ---------------- fused SAGE layer: out = [relu]( mean(agg) @ Wl + b + x @ Wr ) ----
// All operands bf16; weights pre-transposed ([col][k] contiguous in k).
// Block = 256 threads = 8 waves; block handles 16 rows; wave w handles cols [16w,16w+16).
__global__ __launch_bounds__(256)
void sage_gemm_kernel(const __bf16* __restrict__ aggb,
                      const __bf16* __restrict__ xb,
                      const __bf16* __restrict__ WTl,
                      const __bf16* __restrict__ WTr,
                      const float* __restrict__ bias,
                      __bf16* __restrict__ out,
                      int relu) {
    const int wv   = threadIdx.x >> 5;     // n-tile 0..7
    const int lane = threadIdx.x & 31;
    const int half = lane >> 4;
    const int r    = lane & 15;
    const int mrow = blockIdx.x * 16 + r;  // A row for this lane (both halves)
    const int col  = wv * 16 + r;          // B/C column for this lane

    const __bf16* __restrict__ arow = aggb + (size_t)mrow * FDIM;
    const __bf16* __restrict__ xrow = xb   + (size_t)mrow * FDIM;
    const __bf16* __restrict__ wl   = WTl  + (size_t)col * FDIM;
    const __bf16* __restrict__ wr   = WTr  + (size_t)col * FDIM;

    v8f acc = {};
#pragma unroll
    for (int kb = 0; kb < FDIM; kb += 32) {
        const int k0 = kb + half * 16;      // first 8-elem chunk
        const int k1 = k0 + 8;              // second 8-elem chunk
        V16 aa, ax;
        aa.h[0] = *(const bf16x8*)(arow + k0);
        aa.h[1] = *(const bf16x8*)(arow + k1);
        ax.h[0] = *(const bf16x8*)(xrow + k0);
        ax.h[1] = *(const bf16x8*)(xrow + k1);
        const int kB = kb + half * 16;      // 16 contiguous k for this lane's column
        const v16bf bl = *(const v16bf*)(wl + kB);
        const v16bf br = *(const v16bf*)(wr + kB);
        acc = __builtin_amdgcn_wmma_f32_16x16x32_bf16(false, aa.v, false, bl,
                                                      (short)0, acc, false, false);
        acc = __builtin_amdgcn_wmma_f32_16x16x32_bf16(false, ax.v, false, br,
                                                      (short)0, acc, false, false);
    }

    const float b = bias[col];
#pragma unroll
    for (int v = 0; v < 8; ++v) {
        float val = acc[v] + b;
        if (relu) val = fmaxf(val, 0.0f);
        out[(size_t)(blockIdx.x * 16 + v + half * 8) * HDIM + col] = (__bf16)val;
    }
}

// ---------------- decoder: sigmoid( relu(concat(z_rna[lr], z_dis[lc]) @ dW1 + db1) @ dW2 + db2 )
// Block = 256 threads = 8 waves; block handles 16 labels; wave w covers hidden cols [16w,16w+16).
__global__ __launch_bounds__(256)
void decoder_kernel(const __bf16* __restrict__ zb_rna,
                    const __bf16* __restrict__ zb_dis,
                    const int* __restrict__ lrow,
                    const int* __restrict__ lcol,
                    const __bf16* __restrict__ dW1T,   // bf16 [128][256]
                    const float* __restrict__ db1,
                    const float* __restrict__ dW2,
                    const float* __restrict__ db2,
                    float* __restrict__ out) {
    __shared__ float acc16[16];
    const int wv   = threadIdx.x >> 5;
    const int lane = threadIdx.x & 31;
    const int half = lane >> 4;
    const int r    = lane & 15;
    const int label = blockIdx.x * 16 + r;
    const int lr = lrow[label];
    const int lc = lcol[label];
    const __bf16* __restrict__ zr = zb_rna + (size_t)lr * HDIM;
    const __bf16* __restrict__ zd = zb_dis + (size_t)lc * HDIM;
    const __bf16* __restrict__ w1 = dW1T + (size_t)(wv * 16 + r) * (2 * HDIM);

    if (threadIdx.x < 16) acc16[threadIdx.x] = 0.0f;
    __syncthreads();

    const int col = wv * 16 + r;
    v8f c = {};
#pragma unroll
    for (int kb = 0; kb < 2 * HDIM; kb += 32) {
        const __bf16* srow = (kb < HDIM) ? (zr + kb) : (zd + (kb - HDIM));
        V16 a;
        a.h[0] = *(const bf16x8*)(srow + half * 16);
        a.h[1] = *(const bf16x8*)(srow + 8 + half * 16);
        const v16bf b = *(const v16bf*)(w1 + kb + half * 16);
        c = __builtin_amdgcn_wmma_f32_16x16x32_bf16(false, a.v, false, b,
                                                    (short)0, c, false, false);
    }

    const float w2 = dW2[col];
    const float b1 = db1[col];
#pragma unroll
    for (int v = 0; v < 8; ++v) {
        float hv = fmaxf(c[v] + b1, 0.0f);
        atomicAdd(&acc16[v + half * 8], hv * w2);   // ds_add_f32
    }
    __syncthreads();

    if (threadIdx.x < 16) {
        float t = acc16[threadIdx.x] + db2[0];
        out[blockIdx.x * 16 + threadIdx.x] = 1.0f / (1.0f + expf(-t));
    }
}

// ---------------- host launcher ----------------
extern "C" void kernel_launch(void* const* d_in, const int* in_sizes, int n_in,
                              void* d_out, int out_size, void* d_ws, size_t ws_size,
                              hipStream_t stream) {
    const float* x_rna = (const float*)d_in[0];
    const float* x_dis = (const float*)d_in[1];
    const int* esrc_rd = (const int*)d_in[2];
    const int* edst_rd = (const int*)d_in[3];
    const int* esrc_dr = (const int*)d_in[4];
    const int* edst_dr = (const int*)d_in[5];
    const int* lrow    = (const int*)d_in[6];
    const int* lcol    = (const int*)d_in[7];
    const float* W1rd_l = (const float*)d_in[8];
    const float* b1rd   = (const float*)d_in[9];
    const float* W1rd_r = (const float*)d_in[10];
    const float* W1dr_l = (const float*)d_in[11];
    const float* b1dr   = (const float*)d_in[12];
    const float* W1dr_r = (const float*)d_in[13];
    const float* W2rd_l = (const float*)d_in[14];
    const float* b2rd   = (const float*)d_in[15];
    const float* W2rd_r = (const float*)d_in[16];
    const float* W2dr_l = (const float*)d_in[17];
    const float* b2dr   = (const float*)d_in[18];
    const float* W2dr_r = (const float*)d_in[19];
    const float* dW1    = (const float*)d_in[20];
    const float* db1    = (const float*)d_in[21];
    const float* dW2    = (const float*)d_in[22];
    const float* db2    = (const float*)d_in[23];
    float* out = (float*)d_out;

    char* ws = (char*)d_ws;
    float*  agg_dis  = (float*)(ws + B_AGG_DIS);
    float*  cnt_dis  = (float*)(ws + B_CNT_DIS);
    float*  agg_rna  = (float*)(ws + B_AGG_RNA);
    float*  cnt_rna  = (float*)(ws + B_CNT_RNA);
    __bf16* xb_rna   = (__bf16*)(ws + B_XB_RNA);
    __bf16* xb_dis   = (__bf16*)(ws + B_XB_DIS);
    __bf16* aggb_dis = (__bf16*)(ws + B_AGGB_DIS);
    __bf16* aggb_rna = (__bf16*)(ws + B_AGGB_RNA);
    __bf16* h_dis    = (__bf16*)(ws + B_H_DIS);
    __bf16* h_rna    = (__bf16*)(ws + B_H_RNA);
    __bf16* z_dis    = (__bf16*)(ws + B_Z_DIS);
    __bf16* z_rna    = (__bf16*)(ws + B_Z_RNA);
    __bf16* WT       = (__bf16*)(ws + B_WT);     // 8 matrices of 128*128
    __bf16* W1rdTl = WT + 0 * 128 * 128;
    __bf16* W1rdTr = WT + 1 * 128 * 128;
    __bf16* W1drTl = WT + 2 * 128 * 128;
    __bf16* W1drTr = WT + 3 * 128 * 128;
    __bf16* W2rdTl = WT + 4 * 128 * 128;
    __bf16* W2rdTr = WT + 5 * 128 * 128;
    __bf16* W2drTl = WT + 6 * 128 * 128;
    __bf16* W2drTr = WT + 7 * 128 * 128;
    __bf16* dW1T   = (__bf16*)(ws + B_DW1T);

    const int scatter_blocks = NEDGE / 8;   // 8 waves per 256-thread block, 1 edge/wave
    const int nzero = (int)(B_ZERO_END / 4);

    // ---- one-time conversions (cheap, re-run every launch for determinism) ----
    transpose_cvt_kernel<<<64, 256, 0, stream>>>(W1rd_l, W1rdTl, 128);
    transpose_cvt_kernel<<<64, 256, 0, stream>>>(W1rd_r, W1rdTr, 128);
    transpose_cvt_kernel<<<64, 256, 0, stream>>>(W1dr_l, W1drTl, 128);
    transpose_cvt_kernel<<<64, 256, 0, stream>>>(W1dr_r, W1drTr, 128);
    transpose_cvt_kernel<<<64, 256, 0, stream>>>(W2rd_l, W2rdTl, 128);
    transpose_cvt_kernel<<<64, 256, 0, stream>>>(W2rd_r, W2rdTr, 128);
    transpose_cvt_kernel<<<64, 256, 0, stream>>>(W2dr_l, W2drTl, 128);
    transpose_cvt_kernel<<<64, 256, 0, stream>>>(W2dr_r, W2drTr, 128);
    transpose_cvt_kernel<<<128, 256, 0, stream>>>(dW1, dW1T, 256);
    cvt_bf16_kernel<<<8192, 256, 0, stream>>>(x_rna, xb_rna, N_RNA * FDIM);
    cvt_bf16_kernel<<<2048, 256, 0, stream>>>(x_dis, xb_dis, N_DIS * FDIM);

    // ---- layer 1 ----
    zero_kernel<<<4096, 256, 0, stream>>>((float*)ws, nzero);
    scatter_kernel<<<scatter_blocks, 256, 0, stream>>>(xb_rna, esrc_rd, edst_rd,
                                                       agg_dis, cnt_dis, NEDGE);
    scatter_kernel<<<scatter_blocks, 256, 0, stream>>>(xb_dis, esrc_dr, edst_dr,
                                                       agg_rna, cnt_rna, NEDGE);
    finalize_mean_kernel<<<2048, 256, 0, stream>>>(agg_dis, cnt_dis, aggb_dis, N_DIS * FDIM);
    finalize_mean_kernel<<<8192, 256, 0, stream>>>(agg_rna, cnt_rna, aggb_rna, N_RNA * FDIM);
    sage_gemm_kernel<<<N_DIS / 16, 256, 0, stream>>>(aggb_dis, xb_dis,
                                                     W1rdTl, W1rdTr, b1rd, h_dis, 1);
    sage_gemm_kernel<<<N_RNA / 16, 256, 0, stream>>>(aggb_rna, xb_rna,
                                                     W1drTl, W1drTr, b1dr, h_rna, 1);

    // ---- layer 2 ----
    zero_kernel<<<4096, 256, 0, stream>>>((float*)ws, nzero);
    scatter_kernel<<<scatter_blocks, 256, 0, stream>>>(h_rna, esrc_rd, edst_rd,
                                                       agg_dis, cnt_dis, NEDGE);
    scatter_kernel<<<scatter_blocks, 256, 0, stream>>>(h_dis, esrc_dr, edst_dr,
                                                       agg_rna, cnt_rna, NEDGE);
    finalize_mean_kernel<<<2048, 256, 0, stream>>>(agg_dis, cnt_dis, aggb_dis, N_DIS * FDIM);
    finalize_mean_kernel<<<8192, 256, 0, stream>>>(agg_rna, cnt_rna, aggb_rna, N_RNA * FDIM);
    sage_gemm_kernel<<<N_DIS / 16, 256, 0, stream>>>(aggb_dis, h_dis,
                                                     W2rdTl, W2rdTr, b2rd, z_dis, 0);
    sage_gemm_kernel<<<N_RNA / 16, 256, 0, stream>>>(aggb_rna, h_rna,
                                                     W2drTl, W2drTr, b2dr, z_rna, 0);

    // ---- decoder ----
    decoder_kernel<<<NLAB / 16, 256, 0, stream>>>(z_rna, z_dis, lrow, lcol,
                                                  dW1T, db1, dW2, db2, out);
}
